// Attention_41704132444382
// MI455X (gfx1250) — matrix operand order repaired and checked by reading, not compiled, not run on hardware
//
#include <hip/hip_runtime.h>
#include <hip/hip_bf16.h>

typedef _Float16 h16;
typedef __attribute__((ext_vector_type(16))) _Float16 v16h;
typedef __attribute__((ext_vector_type(8)))  _Float16 v8h;
typedef __attribute__((ext_vector_type(8)))  float    v8f;
typedef __attribute__((ext_vector_type(8)))  unsigned u32x8;

#define DIM    1024
#define BATCH  2
#define SEQ    2048
#define HEADS  16
#define DH     64
#define MROWS  (BATCH * SEQ)        /* 4096 */
#define QK_SCALE 0.03125f           /* 1024^-0.5 */

// ---------------------------------------------------------------------------
// Fragment helpers (contiguous k order k=16*half+i, wave32)
// ---------------------------------------------------------------------------
static __device__ __forceinline__ v16h ld_frag(const h16* p0, const h16* p1) {
  v8h lo = *(const v8h*)p0;
  v8h hi = *(const v8h*)p1;
  return __builtin_shufflevector(lo, hi, 0,1,2,3,4,5,6,7,8,9,10,11,12,13,14,15);
}

// A-matrix 16x32: lane holds row m=lane%16; elems i -> K=half*16+i.
// Row-major source => two b128 loads.
static __device__ __forceinline__ v16h a_frag(const h16* rowp, int k, int half) {
  return ld_frag(rowp + k + half * 16, rowp + k + half * 16 + 8);
}

// B-matrix 32x16: lane holds col n=lane%16; elems i -> K=half*16+i.
// Source = row n of W (row-major, K contiguous) => two b128 loads.
static __device__ __forceinline__ v16h b_frag(const h16* rowp, int k, int half) {
  return ld_frag(rowp + k + half * 16, rowp + k + half * 16 + 8);
}

static __device__ __forceinline__ v8f wmma(v16h a, v16h b, v8f c) {
  return __builtin_amdgcn_wmma_f32_16x16x32_f16(false, a, false, b,
                                                (short)0, c, false, false);
}

static __device__ __forceinline__ unsigned pkh(float a, float b) {
  unsigned short ua = __builtin_bit_cast(unsigned short, (h16)a);
  unsigned short ub = __builtin_bit_cast(unsigned short, (h16)b);
  return (unsigned)ua | ((unsigned)ub << 16);
}

// ---------------------------------------------------------------------------
// Prep: gate (sigmoid(m)>0.5 <=> m>0) + f32->f16 conversion
// ---------------------------------------------------------------------------
__global__ void gate_cvt_kernel(const float* __restrict__ w,
                                const float* __restrict__ m,
                                h16* __restrict__ o, int n) {
  for (int i = blockIdx.x * blockDim.x + threadIdx.x; i < n;
       i += gridDim.x * blockDim.x)
    o[i] = (m[i] > 0.f) ? (h16)w[i] : (h16)0.f;
}

__global__ void cvt_kernel(const float* __restrict__ x, h16* __restrict__ o, int n) {
  for (int i = blockIdx.x * blockDim.x + threadIdx.x; i < n;
       i += gridDim.x * blockDim.x)
    o[i] = (h16)x[i];
}

__global__ void bias_gate_kernel(const float* __restrict__ b,
                                 const float* __restrict__ m,
                                 float* __restrict__ o, int n) {
  for (int i = blockIdx.x * blockDim.x + threadIdx.x; i < n;
       i += gridDim.x * blockDim.x)
    o[i] = (m[i] > 0.f) ? b[i] : 0.f;
}

// ---------------------------------------------------------------------------
// QKV GEMM: [4096,1024] x [3072,1024]^T, one wave -> 32x64 tile
// (two 16-row A-frags reuse the four B-frags -> 8 WMMA / 12 b128 per k-step).
// Writes Qh (scaled), Kh as [bh][n][d] f16 and V transposed Vt[bh][d][n] f16.
// ---------------------------------------------------------------------------
__global__ __launch_bounds__(32)
void qkv_gemm_kernel(const h16* __restrict__ Xh, const h16* __restrict__ Wh,
                     h16* __restrict__ Qh, h16* __restrict__ Kh,
                     h16* __restrict__ Vt) {
  const int lane = threadIdx.x & 31;
  const int half = lane >> 4;
  const int li   = lane & 15;
  const int mb   = blockIdx.x & 127;   // 128 M-tiles of 32
  const int nb   = blockIdx.x >> 7;    // 48 N-blocks of 64
  const int m0   = mb * 32;
  const int n0   = nb * 64;

  const h16* arow0 = Xh + (size_t)(m0 +  0 + li) * DIM;
  const h16* arow1 = Xh + (size_t)(m0 + 16 + li) * DIM;
  const h16* br0   = Wh + (size_t)(n0 +  0 + li) * DIM;
  const h16* br1   = Wh + (size_t)(n0 + 16 + li) * DIM;
  const h16* br2   = Wh + (size_t)(n0 + 32 + li) * DIM;
  const h16* br3   = Wh + (size_t)(n0 + 48 + li) * DIM;

  v8f acc[2][4] = {{v8f{}, v8f{}, v8f{}, v8f{}}, {v8f{}, v8f{}, v8f{}, v8f{}}};
#pragma unroll 2
  for (int k = 0; k < DIM; k += 32) {
    __builtin_prefetch(arow0 + k + 128, 0, 1);
    v16h a0 = a_frag(arow0, k, half);
    v16h a1 = a_frag(arow1, k, half);
    v16h b0 = b_frag(br0, k, half);
    v16h b1 = b_frag(br1, k, half);
    v16h b2 = b_frag(br2, k, half);
    v16h b3 = b_frag(br3, k, half);
    acc[0][0] = wmma(a0, b0, acc[0][0]);
    acc[1][0] = wmma(a1, b0, acc[1][0]);
    acc[0][1] = wmma(a0, b1, acc[0][1]);
    acc[1][1] = wmma(a1, b1, acc[1][1]);
    acc[0][2] = wmma(a0, b2, acc[0][2]);
    acc[1][2] = wmma(a1, b2, acc[1][2]);
    acc[0][3] = wmma(a0, b3, acc[0][3]);
    acc[1][3] = wmma(a1, b3, acc[1][3]);
  }

  const int seg  = n0 >> 10;               // 0=Q 1=K 2=V
  const int head = (n0 & 1023) >> 6;
  const size_t bh = (size_t)((m0 >> 11) * HEADS + head);

  if (seg < 2) {
    const float sc = (seg == 0) ? QK_SCALE : 1.f;
#pragma unroll
    for (int g = 0; g < 2; ++g) {
      const int tok0 = (m0 & (SEQ - 1)) + g * 16;
      h16* base = (seg == 0 ? Qh : Kh) + (bh * SEQ + tok0) * DH;
#pragma unroll
      for (int c = 0; c < 4; ++c) {
        h16* p = base + c * 16 + li;
#pragma unroll
        for (int r = 0; r < 8; ++r)
          p[(size_t)(r + 8 * half) * DH] = (h16)(acc[g][c][r] * sc);
      }
    }
  } else {
#pragma unroll
    for (int g = 0; g < 2; ++g) {
      const int tok0 = (m0 & (SEQ - 1)) + g * 16;
#pragma unroll
      for (int c = 0; c < 4; ++c) {
        v8h pk;
#pragma unroll
        for (int r = 0; r < 8; ++r) pk[r] = (h16)acc[g][c][r];
        h16* p = Vt + (bh * DH + (size_t)(c * 16 + li)) * SEQ + tok0 + 8 * half;
        *(v8h*)p = pk;
      }
    }
  }
}

// ---------------------------------------------------------------------------
// Flash attention, one wave per (bh, 32-query tile). Computes S^T = K*Q^T so
// softmax stats are per-lane; P^T -> B-frag via shfl_xor(16); O^T = Vt * P^T.
// K/V fragments are loaded once and reused by both 16-query groups.
// ---------------------------------------------------------------------------
__global__ __launch_bounds__(32)
void attn_kernel(const h16* __restrict__ Qh, const h16* __restrict__ Kh,
                 const h16* __restrict__ Vt, h16* __restrict__ Oh) {
  const int lane = threadIdx.x & 31;
  const int half = lane >> 4;
  const int li   = lane & 15;
  const int qt   = blockIdx.x & 63;   // 64 tiles of 32 queries
  const int bh   = blockIdx.x >> 6;   // 0..31
  const int q0   = qt * 32;

  v16h bq[2][2];
#pragma unroll
  for (int g = 0; g < 2; ++g) {
    const h16* qrow = Qh + ((size_t)bh * SEQ + q0 + g * 16 + li) * DH;
    bq[g][0] = b_frag(qrow, 0, half);    // d = 0..31
    bq[g][1] = b_frag(qrow, 32, half);   // d = 32..63
  }

  const h16* kbase = Kh + (size_t)bh * SEQ * DH;
  const h16* vbase = Vt + (size_t)bh * DH * SEQ;

  float mrow[2] = {-1e30f, -1e30f};
  float lrow[2] = {0.f, 0.f};
  v8f oacc[2][4] = {{v8f{}, v8f{}, v8f{}, v8f{}}, {v8f{}, v8f{}, v8f{}, v8f{}}};

#pragma unroll 1
  for (int jb = 0; jb < SEQ; jb += 32) {
    const h16* kr0 = kbase + (size_t)(jb + li) * DH;
    const h16* kr1 = kbase + (size_t)(jb + 16 + li) * DH;
    const v16h ak00 = a_frag(kr0, 0, half);   // keys jb..jb+15, d 0..31
    const v16h ak01 = a_frag(kr0, 32, half);  // keys jb..jb+15, d 32..63
    const v16h ak10 = a_frag(kr1, 0, half);
    const v16h ak11 = a_frag(kr1, 32, half);

    v16h pb[2];
#pragma unroll
    for (int g = 0; g < 2; ++g) {
      v8f s0 = {}, s1 = {};
      s0 = wmma(ak00, bq[g][0], s0);
      s0 = wmma(ak01, bq[g][1], s0);
      s1 = wmma(ak10, bq[g][0], s1);
      s1 = wmma(ak11, bq[g][1], s1);

      // per-query block max (regs + partner half-lane)
      float bm = s0[0];
#pragma unroll
      for (int r = 1; r < 8; ++r) bm = fmaxf(bm, s0[r]);
#pragma unroll
      for (int r = 0; r < 8; ++r) bm = fmaxf(bm, s1[r]);
      bm = fmaxf(bm, __shfl_xor(bm, 16, 32));

      const float mnew  = fmaxf(mrow[g], bm);
      const float alpha = __expf(mrow[g] - mnew);
      mrow[g] = mnew;

      float ps = 0.f;
#pragma unroll
      for (int r = 0; r < 8; ++r) { s0[r] = __expf(s0[r] - mnew); ps += s0[r]; }
#pragma unroll
      for (int r = 0; r < 8; ++r) { s1[r] = __expf(s1[r] - mnew); ps += s1[r]; }
      ps += __shfl_xor(ps, 16, 32);
      lrow[g] = lrow[g] * alpha + ps;

#pragma unroll
      for (int c = 0; c < 4; ++c)
#pragma unroll
        for (int r = 0; r < 8; ++r) oacc[g][c][r] *= alpha;

      // P^T (32 keys x 16 queries) -> B fragment: exchange with lane^16
      u32x8 bb;
#pragma unroll
      for (int j = 0; j < 4; ++j) {
        unsigned u0 = pkh(s0[2 * j], s0[2 * j + 1]);   // own kb0, j_local=2j+8*half
        unsigned u1 = pkh(s1[2 * j], s1[2 * j + 1]);   // own kb1
        unsigned w0 = (unsigned)__shfl_xor((int)u0, 16, 32);
        unsigned w1 = (unsigned)__shfl_xor((int)u1, 16, 32);
        bb[j]     = half ? w1 : u0;   // K elems 2j, 2j+1
        bb[4 + j] = half ? u1 : w0;   // K elems 8+2j, 9+2j
      }
      pb[g] = __builtin_bit_cast(v16h, bb);
    }

    // O^T += Vt-rows * P^T ; V fragments shared by both query groups
#pragma unroll
    for (int c = 0; c < 4; ++c) {
      const h16* vrow = vbase + (size_t)(c * 16 + li) * SEQ + jb;
      const v16h av = a_frag(vrow, 0, half);
      oacc[0][c] = wmma(av, pb[0], oacc[0][c]);
      oacc[1][c] = wmma(av, pb[1], oacc[1][c]);
    }
  }

#pragma unroll
  for (int g = 0; g < 2; ++g) {
    const float inv = 1.f / lrow[g];
    h16* obase = Oh + ((size_t)(bh >> 4) * SEQ + q0 + g * 16 + li) * DIM
                    + (size_t)(bh & 15) * DH;
#pragma unroll
    for (int c = 0; c < 4; ++c) {
      v8h pk;
#pragma unroll
      for (int r = 0; r < 8; ++r) pk[r] = (h16)(oacc[g][c][r] * inv);
      *(v8h*)(obase + c * 16 + 8 * half) = pk;
    }
  }
}

// ---------------------------------------------------------------------------
// Output projection: [4096,1024] x [1024,1024]^T + gated bias, f32 out.
// One wave -> 32x64 tile.
// ---------------------------------------------------------------------------
__global__ __launch_bounds__(32)
void out_gemm_kernel(const h16* __restrict__ Oh, const h16* __restrict__ Wh,
                     const float* __restrict__ biasg, float* __restrict__ out) {
  const int lane = threadIdx.x & 31;
  const int half = lane >> 4;
  const int li   = lane & 15;
  const int mb   = blockIdx.x & 127;   // 128 M-tiles of 32
  const int nb   = blockIdx.x >> 7;    // 16 N-blocks of 64
  const int m0   = mb * 32;
  const int n0   = nb * 64;

  const h16* arow0 = Oh + (size_t)(m0 +  0 + li) * DIM;
  const h16* arow1 = Oh + (size_t)(m0 + 16 + li) * DIM;
  const h16* br0   = Wh + (size_t)(n0 +  0 + li) * DIM;
  const h16* br1   = Wh + (size_t)(n0 + 16 + li) * DIM;
  const h16* br2   = Wh + (size_t)(n0 + 32 + li) * DIM;
  const h16* br3   = Wh + (size_t)(n0 + 48 + li) * DIM;

  v8f acc[2][4] = {{v8f{}, v8f{}, v8f{}, v8f{}}, {v8f{}, v8f{}, v8f{}, v8f{}}};
#pragma unroll 2
  for (int k = 0; k < DIM; k += 32) {
    __builtin_prefetch(arow0 + k + 128, 0, 1);
    v16h a0 = a_frag(arow0, k, half);
    v16h a1 = a_frag(arow1, k, half);
    v16h b0 = b_frag(br0, k, half);
    v16h b1 = b_frag(br1, k, half);
    v16h b2 = b_frag(br2, k, half);
    v16h b3 = b_frag(br3, k, half);
    acc[0][0] = wmma(a0, b0, acc[0][0]);
    acc[1][0] = wmma(a1, b0, acc[1][0]);
    acc[0][1] = wmma(a0, b1, acc[0][1]);
    acc[1][1] = wmma(a1, b1, acc[1][1]);
    acc[0][2] = wmma(a0, b2, acc[0][2]);
    acc[1][2] = wmma(a1, b2, acc[1][2]);
    acc[0][3] = wmma(a0, b3, acc[0][3]);
    acc[1][3] = wmma(a1, b3, acc[1][3]);
  }

#pragma unroll
  for (int c = 0; c < 4; ++c) {
    const float bv = biasg[n0 + c * 16 + li];
#pragma unroll
    for (int g = 0; g < 2; ++g) {
      float* ob = out + (size_t)(m0 + g * 16) * DIM + n0 + c * 16 + li;
#pragma unroll
      for (int r = 0; r < 8; ++r)
        ob[(size_t)(r + 8 * half) * DIM] = acc[g][c][r] + bv;
    }
  }
}

// ---------------------------------------------------------------------------
// Host launcher
// ---------------------------------------------------------------------------
extern "C" void kernel_launch(void* const* d_in, const int* in_sizes, int n_in,
                              void* d_out, int out_size, void* d_ws, size_t ws_size,
                              hipStream_t stream) {
  const float* x       = (const float*)d_in[0];
  const float* wqkv    = (const float*)d_in[1];
  const float* wqkv_m  = (const float*)d_in[2];
  const float* wout    = (const float*)d_in[3];
  const float* wout_m  = (const float*)d_in[4];
  const float* bias    = (const float*)d_in[5];
  const float* bias_m  = (const float*)d_in[6];
  float* out = (float*)d_out;

  char* ws = (char*)d_ws;
  size_t off = 0;
  auto carve = [&](size_t bytes) -> void* {
    void* p = ws + off;
    off = (off + bytes + 255) & ~(size_t)255;
    return p;
  };
  h16*   Xh     = (h16*)carve((size_t)MROWS * DIM * 2);       // 8 MiB
  h16*   Wqkvh  = (h16*)carve((size_t)3 * DIM * DIM * 2);     // 6 MiB
  h16*   Wouth  = (h16*)carve((size_t)DIM * DIM * 2);         // 2 MiB
  float* biasg  = (float*)carve((size_t)DIM * 4);
  h16*   Qh     = (h16*)carve((size_t)BATCH * HEADS * SEQ * DH * 2); // 8 MiB
  h16*   Kh     = (h16*)carve((size_t)BATCH * HEADS * SEQ * DH * 2); // 8 MiB
  h16*   Vt     = (h16*)carve((size_t)BATCH * HEADS * SEQ * DH * 2); // 8 MiB
  h16*   Oh     = (h16*)carve((size_t)MROWS * DIM * 2);       // 8 MiB

  gate_cvt_kernel<<<4096, 256, 0, stream>>>(wqkv, wqkv_m, Wqkvh, 3 * DIM * DIM);
  gate_cvt_kernel<<<2048, 256, 0, stream>>>(wout, wout_m, Wouth, DIM * DIM);
  cvt_kernel<<<4096, 256, 0, stream>>>(x, Xh, MROWS * DIM);
  bias_gate_kernel<<<4, 256, 0, stream>>>(bias, bias_m, biasg, DIM);

  qkv_gemm_kernel<<<128 * 48, 32, 0, stream>>>(Xh, Wqkvh, Qh, Kh, Vt);
  attn_kernel<<<BATCH * HEADS * (SEQ / 32), 32, 0, stream>>>(Qh, Kh, Vt, Oh);
  out_gemm_kernel<<<128 * 16, 32, 0, stream>>>(Oh, Wouth, biasg, out);
}
